// DCCEngine_292057776507
// MI455X (gfx1250) — matrix operand order repaired and checked by reading, compiled, not run on hardware
//
#include <hip/hip_runtime.h>
#include <hip/hip_bf16.h>
#include <stdint.h>
#include <stddef.h>

typedef __attribute__((ext_vector_type(16))) __bf16 v16bf;
typedef __attribute__((ext_vector_type(16))) unsigned short v16u;
typedef __attribute__((ext_vector_type(8)))  float v8f;

static constexpr int Bn = 8, Cn = 256, HWn = 16384, Kn = 64;
static constexpr int GROUPS = 32;
static constexpr float EPS = 1e-5f;

__device__ __forceinline__ unsigned short f2bf(float f) {
  unsigned u = __builtin_bit_cast(unsigned, f);
  u += 0x7FFFu + ((u >> 16) & 1u);          // round-to-nearest-even
  return (unsigned short)(u >> 16);
}
__device__ __forceinline__ float bf2f(unsigned short h) {
  unsigned u = ((unsigned)h) << 16;
  return __builtin_bit_cast(float, u);
}
// A-fragment (16-bit, 16x32) K index for element j in lane-half lh:
// lanes 0-15: K = {0..15}; lanes 16-31: K = {16..31}
__device__ __forceinline__ int kaIdx(int j, int lh) {
  return 16 * lh + j;
}
__device__ __forceinline__ v8f wmma_bf16(v16u a, v16u b, v8f c) {
  return __builtin_amdgcn_wmma_f32_16x16x32_bf16(
      false, __builtin_bit_cast(v16bf, a),
      false, __builtin_bit_cast(v16bf, b),
      (short)0, c, false, false);
}

__global__ void k_zero(float* p, int n) {
  int i = blockIdx.x * blockDim.x + threadIdx.x;
  if (i < n) p[i] = 0.0f;
}

__global__ void k_cvt(const float* __restrict__ pw, const float* __restrict__ cen,
                      unsigned short* __restrict__ pw_bf, unsigned short* __restrict__ cen_bf) {
  int i = blockIdx.x * blockDim.x + threadIdx.x;
  if (i < Cn * Cn) pw_bf[i] = f2bf(pw[i]);
  if (i < Kn * Cn) cen_bf[i] = f2bf(cen[i]);
}

// ---------------------------------------------------------------------------
// Kernel 1: sim = x_flat . centers^T  (WMMA, bf16), softmax over K=64,
// store attn as bf16 [B][N][64]. 4 waves/block, 16 pixels per wave.
// ---------------------------------------------------------------------------
__global__ __launch_bounds__(128) void k_attn(const float* __restrict__ x,
                                              const unsigned short* __restrict__ cen_bf,
                                              unsigned short* __restrict__ attn_bf) {
  __shared__ unsigned short xt[4][16][Cn];   // per-wave x tile, bf16  (32 KB)
  __shared__ float simbuf[64][65];           // sim staging for softmax (16.6 KB)
  const int b = blockIdx.x >> 8;
  const int pix0 = (blockIdx.x & 255) * 64;
  const int tid = threadIdx.x;
  const int wave = tid >> 5, lane = tid & 31;
  const int lh = lane >> 4, m = lane & 15;
  const int wpix0 = pix0 + wave * 16;

  // stage x tile: x[b][c][pix] -> xt[pixel][c] (bf16)
  for (int it = 0; it < 128; ++it) {
    int c = it * 2 + lh;
    float v = x[((size_t)b * Cn + c) * HWn + wpix0 + m];
    xt[wave][m][c] = f2bf(v);
  }
  __syncthreads();

  v8f acc[4] = {};
  for (int ks = 0; ks < 8; ++ks) {           // K = C = 256 in steps of 32
    const int cb = ks * 32;
    v16u av;
#pragma unroll
    for (int j = 0; j < 16; ++j) av[j] = xt[wave][m][cb + kaIdx(j, lh)];
#pragma unroll
    for (int nt = 0; nt < 4; ++nt) {         // N = 64 clusters = 4 tiles
      const unsigned short* bp = cen_bf + (size_t)(nt * 16 + m) * Cn + cb + 16 * lh;
      v16u bv;
#pragma unroll
      for (int j = 0; j < 16; ++j) bv[j] = bp[j];
      acc[nt] = wmma_bf16(av, bv, acc[nt]);
    }
  }
  const float scale = 0.0625f;               // 1/sqrt(C)
#pragma unroll
  for (int nt = 0; nt < 4; ++nt)
#pragma unroll
    for (int r = 0; r < 8; ++r)
      simbuf[wave * 16 + 8 * lh + r][nt * 16 + m] = acc[nt][r] * scale;
  __syncthreads();

  if (tid < 64) {                            // one thread per pixel row
    float mx = -3.0e38f;
    for (int k = 0; k < Kn; ++k) mx = fmaxf(mx, simbuf[tid][k]);
    float s = 0.f;
    for (int k = 0; k < Kn; ++k) { float e = __expf(simbuf[tid][k] - mx); s += e; simbuf[tid][k] = e; }
    float inv = 1.0f / s;
    unsigned short* arow = attn_bf + ((size_t)b * HWn + pix0 + tid) * Kn;
    for (int k = 0; k < Kn; ++k) arow[k] = f2bf(simbuf[tid][k] * inv);
  }
}

// ---------------------------------------------------------------------------
// Kernel 2: cluster[b][k][c] = sum_n attn[b][n][k] * x_flat[b][n][c]
// split-K GEMM over pixel chunks of 512; fp32 atomic reduction.
// 8 waves: waves cover 4 M-tiles (clusters) x 2 N-halves (channels).
// ---------------------------------------------------------------------------
__global__ __launch_bounds__(256) void k_cluster(const float* __restrict__ x,
                                                 const unsigned short* __restrict__ attn_bf,
                                                 float* __restrict__ cluster) {
  const int b = blockIdx.x >> 5;
  const int pixbase = (blockIdx.x & 31) * 512;
  const int tid = threadIdx.x;
  const int wave = tid >> 5, lane = tid & 31;
  const int lh = lane >> 4, m = lane & 15;
  const int mt = wave & 3, nh = wave >> 2;

  v8f acc[8] = {};
  for (int kb = 0; kb < 16; ++kb) {          // 512 pixels in 32-pixel K steps
    const int pix0 = pixbase + kb * 32;
    // prefetch next K-chunk of attn rows (streamed once, then discarded)
    if (kb < 15)
      __builtin_prefetch(attn_bf + ((size_t)b * HWn + pix0 + 32 + (lane & 31)) * Kn, 0, 1);
    const int cl = mt * 16 + m;
    v16u av;                                 // A = attn^T : [cluster][pixel]
#pragma unroll
    for (int j = 0; j < 16; ++j)
      av[j] = attn_bf[((size_t)b * HWn + pix0 + kaIdx(j, lh)) * Kn + cl];
#pragma unroll
    for (int nt = 0; nt < 8; ++nt) {
      const int c = nh * 128 + nt * 16 + m;
      const float* xp = x + ((size_t)b * Cn + c) * HWn + pix0 + 16 * lh;
      v16u bv;                               // B = x_flat : [pixel][c], contiguous in pixel
#pragma unroll
      for (int j = 0; j < 16; ++j) bv[j] = f2bf(xp[j]);
      acc[nt] = wmma_bf16(av, bv, acc[nt]);
    }
  }
#pragma unroll
  for (int nt = 0; nt < 8; ++nt)
#pragma unroll
    for (int r = 0; r < 8; ++r) {
      int row = mt * 16 + 8 * lh + r;
      int c = nh * 128 + nt * 16 + m;
      atomicAdd(&cluster[((size_t)b * Kn + row) * Cn + c], acc[nt][r]);
    }
}

// ---------------------------------------------------------------------------
// Kernel 3: depthwise 7x7 on the 8x8 cluster grid + bias + SiLU.
// Output stored TRANSPOSED as bf16 [b][c][s] so kernel 4 can stage it into
// LDS with a straight contiguous async copy (no VGPR round trip).
// ---------------------------------------------------------------------------
__global__ void k_dwconv(const float* __restrict__ cluster,
                         const float* __restrict__ dw_w, const float* __restrict__ dw_b,
                         unsigned short* __restrict__ refined_bfT) {
  int idx = blockIdx.x * blockDim.x + threadIdx.x;
  if (idx >= Bn * Kn * Cn) return;
  int c = idx & 255;
  int s = (idx >> 8) & 63;
  int b = idx >> 14;
  int sy = s >> 3, sx = s & 7;
  float acc = dw_b[c];
  for (int dy = 0; dy < 7; ++dy) {
    int iy = sy + dy - 3;
    if (iy < 0 || iy >= 8) continue;
    for (int dx = 0; dx < 7; ++dx) {
      int ix = sx + dx - 3;
      if (ix < 0 || ix >= 8) continue;
      acc += cluster[((size_t)b * Kn + iy * 8 + ix) * Cn + c] * dw_w[c * 49 + dy * 7 + dx];
    }
  }
  float v = acc / (1.0f + __expf(-acc));     // SiLU
  refined_bfT[((size_t)b * Cn + c) * Kn + s] = f2bf(v);
}

// ---------------------------------------------------------------------------
// Kernel 4: fused unpool (attn . refined, K=64) -> LDS ->
//           pointwise GEMM (out . pw_w^T, K=256) + bias;
//           writes y (bf16) and accumulates GroupNorm sum / sumsq.
// refined^T tile (32 KB) staged via CDNA5 async global->LDS DMA.
// 2 waves/block, 16 pixels per wave.
// ---------------------------------------------------------------------------
__global__ __launch_bounds__(64) void k_unpool_pw(const unsigned short* __restrict__ attn_bf,
                                                  const unsigned short* __restrict__ refined_bfT,
                                                  const unsigned short* __restrict__ pw_bf,
                                                  const float* __restrict__ pw_b,
                                                  unsigned short* __restrict__ y_bf,
                                                  float* __restrict__ gstats) {
  __shared__ unsigned short refT[Cn][Kn];    // refined transposed [c][k]   (32 KB)
  __shared__ unsigned short outT[2][16][Cn]; // unpool result [wave][pix][c] (16 KB)
  __shared__ float sstat[64];                // per-group {sum, sumsq}
  const int b = blockIdx.x >> 9;
  const int pix0 = (blockIdx.x & 511) * 32;
  const int tid = threadIdx.x;
  const int wave = tid >> 5, lane = tid & 31;
  const int lh = lane >> 4, m = lane & 15;
  const int wpix0 = pix0 + wave * 16;

  // ---- async DMA: refined_bfT[b] (contiguous 32 KB) -> refT in LDS.
  // Low 32 bits of a generic LDS pointer are the LDS byte offset (ISA 10.2).
  {
    const unsigned long long gbase =
        (unsigned long long)(uintptr_t)(refined_bfT + (size_t)b * Cn * Kn);
    const unsigned lbase = (unsigned)(uintptr_t)(&refT[0][0]);
#pragma unroll 4
    for (int i = 0; i < 32; ++i) {           // 64 lanes x 16B x 32 = 32 KB
      unsigned long long ga = gbase + (unsigned long long)(i * 64 + tid) * 16u;
      unsigned la = lbase + (unsigned)((i * 64 + tid) * 16);
      asm volatile("global_load_async_to_lds_b128 %0, %1, off"
                   :: "v"(la), "v"(ga) : "memory");
    }
  }
  sstat[tid] = 0.0f;                          // overlaps with the DMA
  asm volatile("s_wait_asynccnt 0x0" ::: "memory");
  __syncthreads();

  // unpool: out[pix][c] = sum_k attn[pix][k] * refined[k][c]
  v8f acc[16] = {};
#pragma unroll
  for (int kb = 0; kb < 2; ++kb) {           // K = 64 clusters, 2 steps
    const unsigned short* ar = attn_bf + ((size_t)b * HWn + wpix0 + m) * Kn + kb * 32;
    v16u av;
#pragma unroll
    for (int j = 0; j < 16; ++j) av[j] = ar[kaIdx(j, lh)];
#pragma unroll
    for (int nt = 0; nt < 16; ++nt) {        // N = 256 channels
      v16u bv;
#pragma unroll
      for (int j = 0; j < 16; ++j) bv[j] = refT[nt * 16 + m][kb * 32 + 16 * lh + j];
      acc[nt] = wmma_bf16(av, bv, acc[nt]);
    }
  }
#pragma unroll
  for (int nt = 0; nt < 16; ++nt)
#pragma unroll
    for (int r = 0; r < 8; ++r)
      outT[wave][8 * lh + r][nt * 16 + m] = f2bf(acc[nt][r]);
  __syncthreads();

  // pointwise: y[pix][o] = sum_c out[pix][c] * pw_w[o][c]
  v8f acc2[16] = {};
  for (int ks = 0; ks < 8; ++ks) {           // K = C = 256
    const int cb = ks * 32;
    v16u av;
#pragma unroll
    for (int j = 0; j < 16; ++j) av[j] = outT[wave][m][cb + kaIdx(j, lh)];
#pragma unroll
    for (int nt = 0; nt < 16; ++nt) {        // N = 256 outputs
      const unsigned short* wr = pw_bf + (size_t)(nt * 16 + m) * Cn + cb + 16 * lh;
      v16u bv;
#pragma unroll
      for (int j = 0; j < 16; ++j) bv[j] = wr[j];
      acc2[nt] = wmma_bf16(av, bv, acc2[nt]);
    }
  }

#pragma unroll
  for (int nt = 0; nt < 16; ++nt) {
    const int o = nt * 16 + m;
    const float bias = pw_b[o];
    float ls = 0.f, lq = 0.f;
#pragma unroll
    for (int r = 0; r < 8; ++r) {
      int pixel = wpix0 + 8 * lh + r;
      float val = acc2[nt][r] + bias;
      y_bf[((size_t)b * Cn + o) * HWn + pixel] = f2bf(val);
      ls += val; lq += val * val;
    }
    atomicAdd(&sstat[2 * (o >> 3)], ls);
    atomicAdd(&sstat[2 * (o >> 3) + 1], lq);
  }
  __syncthreads();
  atomicAdd(&gstats[(size_t)b * 64 + tid], sstat[tid]);
}

// ---------------------------------------------------------------------------
// Kernel 5: finalize GroupNorm stats, apply GN + residual -> fp32 output
// ---------------------------------------------------------------------------
__global__ __launch_bounds__(256) void k_gn(const float* __restrict__ x,
                                            const unsigned short* __restrict__ y_bf,
                                            const float* __restrict__ gstats,
                                            const float* __restrict__ gn_g,
                                            const float* __restrict__ gn_b,
                                            float* __restrict__ out) {
  size_t idx = (size_t)blockIdx.x * blockDim.x + threadIdx.x;
  if (idx >= (size_t)Bn * Cn * HWn) return;
  int c = (int)((idx / HWn) % Cn);
  int b = (int)(idx / ((size_t)Cn * HWn));
  int g = c >> 3;
  const float cnt = (float)((Cn / GROUPS) * HWn);
  float sum = gstats[b * 64 + 2 * g];
  float sq  = gstats[b * 64 + 2 * g + 1];
  float mean = sum / cnt;
  float var  = sq / cnt - mean * mean;
  float yv = bf2f(y_bf[idx]);
  out[idx] = x[idx] + (yv - mean) * rsqrtf(var + EPS) * gn_g[c] + gn_b[c];
}

extern "C" void kernel_launch(void* const* d_in, const int* in_sizes, int n_in,
                              void* d_out, int out_size, void* d_ws, size_t ws_size,
                              hipStream_t stream) {
  (void)in_sizes; (void)n_in; (void)out_size;
  const float* x       = (const float*)d_in[0];
  const float* centers = (const float*)d_in[1];
  const float* dw_w    = (const float*)d_in[2];
  const float* dw_b    = (const float*)d_in[3];
  const float* pw_w    = (const float*)d_in[4];
  const float* pw_b    = (const float*)d_in[5];
  const float* gn_g    = (const float*)d_in[6];
  const float* gn_b    = (const float*)d_in[7];
  float* out = (float*)d_out;

  char* ws = (char*)d_ws;
  size_t off = 0;
  auto alloc = [&](size_t bytes) { char* p = ws + off; off += (bytes + 255) & ~(size_t)255; return p; };
  unsigned short* attn_bf     = (unsigned short*)alloc((size_t)Bn * HWn * Kn * 2); // 16 MiB
  float*          cluster     = (float*)alloc((size_t)Bn * Kn * Cn * 4);
  unsigned short* refined_bfT = (unsigned short*)alloc((size_t)Bn * Cn * Kn * 2);
  unsigned short* pw_bf       = (unsigned short*)alloc((size_t)Cn * Cn * 2);
  unsigned short* cen_bf      = (unsigned short*)alloc((size_t)Kn * Cn * 2);
  float*          gstats      = (float*)alloc((size_t)Bn * 64 * 4);
  unsigned short* y_bf        = (unsigned short*)alloc((size_t)Bn * Cn * HWn * 2); // 64 MiB
  if (off > ws_size) return;

  k_zero<<<(Bn * Kn * Cn + 255) / 256, 256, 0, stream>>>(cluster, Bn * Kn * Cn);
  k_zero<<<(Bn * 64 + 255) / 256, 256, 0, stream>>>(gstats, Bn * 64);
  k_cvt<<<(Cn * Cn + 255) / 256, 256, 0, stream>>>(pw_w, centers, pw_bf, cen_bf);
  k_attn<<<Bn * (HWn / 64), 128, 0, stream>>>(x, cen_bf, attn_bf);
  k_cluster<<<Bn * (HWn / 512), 256, 0, stream>>>(x, attn_bf, cluster);
  k_dwconv<<<(Bn * Kn * Cn + 255) / 256, 256, 0, stream>>>(cluster, dw_w, dw_b, refined_bfT);
  k_unpool_pw<<<Bn * (HWn / 32), 64, 0, stream>>>(attn_bf, refined_bfT, pw_bf, pw_b, y_bf, gstats);
  k_gn<<<(int)(((size_t)Bn * Cn * HWn + 255) / 256), 256, 0, stream>>>(x, y_bf, gstats, gn_g, gn_b, out);
}
